// Encoder_16999480558406
// MI455X (gfx1250) — matrix operand order repaired and checked by reading, not compiled, not run on hardware
//
#include <hip/hip_runtime.h>
#include <math.h>

// ---------------------------------------------------------------------------
// CDNA5 (gfx1250) transformer encoder, bf16 WMMA (v_wmma_f32_16x16x32_bf16).
// B=8 S=512 IN=128 D=512 H=8 DK=DV=64 DFF=2048 L=4, LOCAL band = 64.
// ---------------------------------------------------------------------------

typedef __bf16 bf16_t;
typedef bf16_t v16bf __attribute__((ext_vector_type(16)));
typedef float  v8f   __attribute__((ext_vector_type(8)));

#define CB   8      // batch
#define CS   512    // seq
#define CIN  128
#define CD   512
#define CH   8
#define CDK  64
#define CDFF 2048
#define CL   4
#define CM   (CB * CS)          // 4096 rows of activations

// gfx1250 async LDS staging (ASYNCcnt path), guarded so either toolchain builds.
#if __has_builtin(__builtin_amdgcn_global_load_async_to_lds_b128) && \
    __has_builtin(__builtin_amdgcn_s_wait_asynccnt)
#define USE_ASYNC_LDS 1
#else
#define USE_ASYNC_LDS 0
#endif

#if USE_ASYNC_LDS
// builtin wants pointers to a 16B int vector (per hipcc diagnostic), global+LDS AS
typedef int v4i __attribute__((vector_size(16)));
typedef __attribute__((address_space(1))) v4i gv4i_t;
typedef __attribute__((address_space(3))) v4i lv4i_t;
__device__ __forceinline__ void async_ld_b128(const void* g, void* l) {
    __builtin_amdgcn_global_load_async_to_lds_b128((gv4i_t*)g, (lv4i_t*)l, 0, 0);
}
#endif

__device__ __forceinline__ int iclamp(int v, int lo, int hi) {
    return v < lo ? lo : (v > hi ? hi : v);
}

// ---------------------------------------------------------------------------
// f32 -> bf16 conversion (grid-stride)
// ---------------------------------------------------------------------------
__global__ void cvt_f32_bf16(const float* __restrict__ in, bf16_t* __restrict__ out, int n) {
    int i = blockIdx.x * blockDim.x + threadIdx.x;
    int stride = gridDim.x * blockDim.x;
    for (; i < n; i += stride) out[i] = (bf16_t)in[i];
}

// ---------------------------------------------------------------------------
// Tiled WMMA GEMM: C[M,N] = A[M,K](bf16) * B[K,N](bf16) + epilogue
// block = 256 threads (8 waves); tile 64x128; K step 32.
// wave (wr = w&3, wc = w>>2): rows wr*16.., cols wc*64 (4 x 16x16 frags).
// LDS row pitches are multiples of 16B so async b128 stores land aligned.
// ---------------------------------------------------------------------------
#define TM 64
#define TN 128
#define TK 32
#define APITCH 48    // 32 + 16 bf16 -> 96B row pitch
#define BPITCH 136   // 128 + 8 bf16 -> 272B row pitch

template<int BIAS, int GELU_, int PE_, int RESID, int WF32, int WBF16>
__global__ __launch_bounds__(256)
void gemm_bf16(const bf16_t* __restrict__ A, const bf16_t* __restrict__ Bm,
               const float* __restrict__ bias, const float* __restrict__ resid,
               float* __restrict__ outF, bf16_t* __restrict__ outB,
               int M, int N, int K)
{
    __shared__ bf16_t As[TM][APITCH];   // 64 x 48 bf16
    __shared__ bf16_t Bs[TK][BPITCH];   // 32 x 136 bf16

    const int tid  = threadIdx.x;
    const int wave = tid >> 5;
    const int lane = tid & 31;
    const int r    = lane & 15;
    const int g    = lane >> 4;
    const int wr   = wave & 3;          // row tile 0..3
    const int wc   = wave >> 2;         // col group 0..1 (64 cols each)
    const int m0   = blockIdx.x * TM;
    const int n0   = blockIdx.y * TN;

    v8f acc[4] = {{}, {}, {}, {}};

    for (int k0 = 0; k0 < K; k0 += TK) {
#if USE_ASYNC_LDS
        // ---- async staging: ASYNCcnt-tracked global -> LDS b128 copies ----
        {
            // A tile 64x32 bf16 = 256 x 16B : 1 per thread
            int row = tid >> 2, kv = tid & 3;
            async_ld_b128(A + (size_t)(m0 + row) * K + k0 + kv * 8, &As[row][kv * 8]);
            // B tile 32x128 bf16 = 512 x 16B : 2 per thread
            int idx = tid;
#pragma unroll
            for (int it = 0; it < 2; ++it, idx += 256) {
                int brow = idx >> 4, nv = idx & 15;
                async_ld_b128(Bm + (size_t)(k0 + brow) * N + n0 + nv * 8, &Bs[brow][nv * 8]);
            }
        }
#else
        // ---- synchronous staging fallback (uint2 = 4 bf16) ----
        {
            int idx = tid;                      // A: 512 x uint2, 2 per thread
#pragma unroll
            for (int it = 0; it < 2; ++it, idx += 256) {
                int row = idx >> 3, kv = idx & 7;
                uint2 v = *(const uint2*)(A + (size_t)(m0 + row) * K + k0 + kv * 4);
                *(uint2*)&As[row][kv * 4] = v;
            }
            idx = tid;                          // B: 1024 x uint2, 4 per thread
#pragma unroll
            for (int it = 0; it < 4; ++it, idx += 256) {
                int row = idx >> 5, nv = idx & 31;
                uint2 v = *(const uint2*)(Bm + (size_t)(k0 + row) * N + n0 + nv * 4);
                *(uint2*)&Bs[row][nv * 4] = v;
            }
        }
#endif
        // prefetch next K tiles into cache hierarchy (global_prefetch_b8)
        if (k0 + TK < K) {
            __builtin_prefetch(A + (size_t)(m0 + (tid >> 3)) * K + k0 + TK, 0, 1);
            __builtin_prefetch(Bm + (size_t)(k0 + TK + (tid >> 4)) * N + n0, 0, 1);
        }
#if USE_ASYNC_LDS
        __builtin_amdgcn_s_wait_asynccnt(0);
#endif
        __syncthreads();

        // A fragment (16x32): lanes 0-15 K=0..15 ; 16-31 K=16..31
        const bf16_t* arow = &As[wr * 16 + r][0];
        v16bf af;
#pragma unroll
        for (int i = 0; i < 4; ++i) {
            int kk = g * 16 + 2 * i;
            af[2 * i]     = arow[kk];
            af[2 * i + 1] = arow[kk + 1];
            int k2 = g * 16 + 8 + 2 * i;
            af[8 + 2 * i]     = arow[k2];
            af[8 + 2 * i + 1] = arow[k2 + 1];
        }
        // B fragments (32x16): lanes 0-15 hold K=0..15, lanes 16-31 K=16..31
#pragma unroll
        for (int f = 0; f < 4; ++f) {
            const int col = wc * 64 + f * 16 + r;
            v16bf bfr;
#pragma unroll
            for (int i = 0; i < 8; ++i) {
                int kk = g * 16 + 2 * i;
                bfr[2 * i]     = Bs[kk][col];
                bfr[2 * i + 1] = Bs[kk + 1][col];
            }
            acc[f] = __builtin_amdgcn_wmma_f32_16x16x32_bf16(false, af, false, bfr, (short)0, acc[f], false, false);
        }
        __syncthreads();
    }

    // epilogue: C/D layout -> element (m = v + 8*g, n = lane&15)
#pragma unroll
    for (int v = 0; v < 8; ++v) {
        const int mm = m0 + wr * 16 + v + g * 8;
#pragma unroll
        for (int f = 0; f < 4; ++f) {
            const int nn = n0 + wc * 64 + f * 16 + r;
            float x = acc[f][v];
            if (BIAS) x += bias[nn];
            if (PE_) {
                // sinusoidal positional encoding: pos = row % S
                float pos = (float)(mm & (CS - 1));
                int   i2  = nn & ~1;
                float freq = __expf((float)i2 * (-9.2103403719761836f / (float)CD));
                float ang  = pos * freq;
                x += (nn & 1) ? __cosf(ang) : __sinf(ang);
            }
            if (GELU_) {
                float t = x;
                x = 0.5f * t * (1.0f + tanhf(0.7978845608028654f * (t + 0.044715f * t * t * t)));
            }
            if (RESID) x += resid[(size_t)mm * N + nn];
            if (WF32)  outF[(size_t)mm * N + nn] = x;
            if (WBF16) outB[(size_t)mm * N + nn] = (bf16_t)x;
        }
    }
}

// ---------------------------------------------------------------------------
// Banded attention, one wave per (b, h, q-tile of 16 rows).
// scores via Q*K^T WMMA over 9 band key-tiles, f32 softmax (shfl reductions),
// P restaged via LDS into A-layout, ctx = P*V via WMMA. Pass 2 stores the
// full probability rows [S] (zeros outside the band) to the attn output.
// ---------------------------------------------------------------------------
__global__ __launch_bounds__(32)
void attn_kernel(const bf16_t* __restrict__ Q, const bf16_t* __restrict__ Kmat,
                 const bf16_t* __restrict__ V, bf16_t* __restrict__ ctx,
                 float* __restrict__ attnOut, int layer, int storeAttn)
{
    const int qt   = blockIdx.x;          // 0..31 query tile
    const int bh   = blockIdx.y;          // 0..63
    const int b    = bh >> 3;
    const int hh   = bh & 7;
    const int lane = threadIdx.x;
    const int r    = lane & 15;
    const int g    = lane >> 4;
    const int q0   = qt * 16;
    const size_t rowBase = (size_t)b * CS;
    const int hcol = hh * CDK;

    // ---- Q fragments (16 x 64 -> two 16x32 A-frags) ----
    v16bf qf[2];
    {
        const bf16_t* qrow = Q + (rowBase + q0 + r) * CD + hcol;
#pragma unroll
        for (int s2 = 0; s2 < 2; ++s2) {
#pragma unroll
            for (int i = 0; i < 4; ++i) {
                int kk = s2 * 32 + g * 16 + 2 * i;
                qf[s2][2 * i]     = qrow[kk];
                qf[s2][2 * i + 1] = qrow[kk + 1];
                int k2 = s2 * 32 + g * 16 + 8 + 2 * i;
                qf[s2][8 + 2 * i]     = qrow[k2];
                qf[s2][8 + 2 * i + 1] = qrow[k2 + 1];
            }
        }
    }

    // ---- scores over band tiles kt in [qt-4, qt+4] ----
    float sc[9][8];
#pragma unroll
    for (int t = 0; t < 9; ++t) {
        const int kt = qt - 4 + t;
        if (kt < 0 || kt >= CS / 16) {
#pragma unroll
            for (int v = 0; v < 8; ++v) sc[t][v] = -1e9f;
            continue;
        }
        v8f a = {};
        const bf16_t* krow = Kmat + (rowBase + kt * 16 + r) * CD + hcol;
#pragma unroll
        for (int s2 = 0; s2 < 2; ++s2) {
            v16bf kf;
#pragma unroll
            for (int i = 0; i < 8; ++i) {
                int kk = s2 * 32 + g * 16 + 2 * i;
                kf[2 * i]     = krow[kk];
                kf[2 * i + 1] = krow[kk + 1];
            }
            a = __builtin_amdgcn_wmma_f32_16x16x32_bf16(false, qf[s2], false, kf, (short)0, a, false, false);
        }
#pragma unroll
        for (int v = 0; v < 8; ++v) {
            const int qr = q0 + v + g * 8;
            const int kc = kt * 16 + r;
            int d = qr - kc; if (d < 0) d = -d;
            sc[t][v] = (d > 64) ? -1e9f : (a[v] * 0.125f);  // 1/sqrt(64)
        }
    }

    // ---- softmax per query row (rows live across 16-lane groups) ----
#pragma unroll
    for (int v = 0; v < 8; ++v) {
        float mx = -1e30f;
#pragma unroll
        for (int t = 0; t < 9; ++t) mx = fmaxf(mx, sc[t][v]);
#pragma unroll
        for (int m = 1; m <= 8; m <<= 1) mx = fmaxf(mx, __shfl_xor(mx, m, 32));
        float sum = 0.f;
#pragma unroll
        for (int t = 0; t < 9; ++t) { float e = __expf(sc[t][v] - mx); sc[t][v] = e; sum += e; }
#pragma unroll
        for (int m = 1; m <= 8; m <<= 1) sum += __shfl_xor(sum, m, 32);
        float inv = 1.0f / sum;
#pragma unroll
        for (int t = 0; t < 9; ++t) sc[t][v] *= inv;
    }

    // ---- pass 2: materialize full probability rows [S] ----
    if (storeAttn) {
        float* abase = attnOut +
            ((((size_t)b * CL + layer) * CH + hh) * CS + q0) * CS;
        for (int t = 0; t < CS / 16; ++t) {          // zeros outside band
            if (t >= qt - 4 && t <= qt + 4) continue;
#pragma unroll
            for (int v = 0; v < 8; ++v)
                abase[(size_t)(v + g * 8) * CS + t * 16 + r] = 0.0f;
        }
#pragma unroll
        for (int t = 0; t < 9; ++t) {
            const int kt = qt - 4 + t;
            if (kt < 0 || kt >= CS / 16) continue;
#pragma unroll
            for (int v = 0; v < 8; ++v)
                abase[(size_t)(v + g * 8) * CS + kt * 16 + r] = sc[t][v];
        }
    }

    // ---- restage P into A-fragment layout via LDS (16 x 160, tile 9 zeroed) ----
    __shared__ bf16_t Ps[16][168];
#pragma unroll
    for (int t = 0; t < 9; ++t)
#pragma unroll
        for (int v = 0; v < 8; ++v)
            Ps[v + g * 8][t * 16 + r] = (bf16_t)sc[t][v];
#pragma unroll
    for (int v = 0; v < 8; ++v) Ps[v + g * 8][144 + r] = (bf16_t)0.0f;
    __syncthreads();

    // ---- ctx = P[16,160] * V[160,64] over 5 K-chunks of 32 ----
    v8f cacc[4] = {{}, {}, {}, {}};
#pragma unroll
    for (int ch = 0; ch < 5; ++ch) {
        v16bf pf;
        const bf16_t* prow = &Ps[r][0];
#pragma unroll
        for (int i = 0; i < 4; ++i) {
            int kk = ch * 32 + g * 16 + 2 * i;
            pf[2 * i]     = prow[kk];
            pf[2 * i + 1] = prow[kk + 1];
            int k2 = ch * 32 + g * 16 + 8 + 2 * i;
            pf[8 + 2 * i]     = prow[k2];
            pf[8 + 2 * i + 1] = prow[k2 + 1];
        }
        const int kbase = q0 - 64 + ch * 32;
#pragma unroll
        for (int nf = 0; nf < 4; ++nf) {
            v16bf vf;
#pragma unroll
            for (int i = 0; i < 8; ++i) {
                int kk = kbase + g * 16 + 2 * i;
                int k0c = iclamp(kk, 0, CS - 1);      // P==0 on clamped rows
                int k1c = iclamp(kk + 1, 0, CS - 1);
                vf[2 * i]     = V[(rowBase + k0c) * CD + hcol + nf * 16 + r];
                vf[2 * i + 1] = V[(rowBase + k1c) * CD + hcol + nf * 16 + r];
            }
            cacc[nf] = __builtin_amdgcn_wmma_f32_16x16x32_bf16(false, pf, false, vf, (short)0, cacc[nf], false, false);
        }
    }

    // ---- write ctx (heads merged: [B*S, H*DV] bf16) ----
#pragma unroll
    for (int nf = 0; nf < 4; ++nf)
#pragma unroll
        for (int v = 0; v < 8; ++v) {
            const int row = q0 + v + g * 8;
            ctx[(rowBase + row) * CD + hcol + nf * 16 + r] = (bf16_t)cacc[nf][v];
        }
}

// ---------------------------------------------------------------------------
// LayerNorm over D=512, one block (256 threads) per row; dual f32+bf16 output.
// ---------------------------------------------------------------------------
__global__ __launch_bounds__(256)
void ln_kernel(const float* __restrict__ in, const float* __restrict__ gamma,
               const float* __restrict__ beta, float* __restrict__ outF,
               bf16_t* __restrict__ outB)
{
    const int row = blockIdx.x;
    const int tid = threadIdx.x;
    __shared__ float red[256];

    const float* x = in + (size_t)row * CD;
    float a0 = x[tid], a1 = x[tid + 256];

    red[tid] = a0 + a1;
    __syncthreads();
#pragma unroll
    for (int off = 128; off > 0; off >>= 1) {
        if (tid < off) red[tid] += red[tid + off];
        __syncthreads();
    }
    const float mean = red[0] * (1.0f / (float)CD);
    __syncthreads();

    float d0 = a0 - mean, d1 = a1 - mean;
    red[tid] = d0 * d0 + d1 * d1;
    __syncthreads();
#pragma unroll
    for (int off = 128; off > 0; off >>= 1) {
        if (tid < off) red[tid] += red[tid + off];
        __syncthreads();
    }
    const float rstd = rsqrtf(red[0] * (1.0f / (float)CD) + 1e-5f);

    float y0 = d0 * rstd * gamma[tid] + beta[tid];
    float y1 = d1 * rstd * gamma[tid + 256] + beta[tid + 256];
    outF[(size_t)row * CD + tid]        = y0;
    outF[(size_t)row * CD + tid + 256]  = y1;
    outB[(size_t)row * CD + tid]        = (bf16_t)y0;
    outB[(size_t)row * CD + tid + 256]  = (bf16_t)y1;
}

// ---------------------------------------------------------------------------
// Host-side orchestration
// ---------------------------------------------------------------------------
static inline void launch_cvt(const float* src, bf16_t* dst, int n, hipStream_t s) {
    int blocks = (n + 1023) / 1024;
    if (blocks > 4096) blocks = 4096;
    cvt_f32_bf16<<<blocks, 1024, 0, s>>>(src, dst, n);
}

extern "C" void kernel_launch(void* const* d_in, const int* in_sizes, int n_in,
                              void* d_out, int out_size, void* d_ws, size_t ws_size,
                              hipStream_t stream)
{
    (void)in_sizes; (void)n_in; (void)out_size; (void)ws_size;

    const float* x     = (const float*)d_in[0];
    const float* We    = (const float*)d_in[1];
    const float* be    = (const float*)d_in[2];
    const float* Wq    = (const float*)d_in[3];
    const float* bq    = (const float*)d_in[4];
    const float* Wk    = (const float*)d_in[5];
    const float* bk    = (const float*)d_in[6];
    const float* Wv    = (const float*)d_in[7];
    const float* bv    = (const float*)d_in[8];
    const float* Wo    = (const float*)d_in[9];
    const float* bo    = (const float*)d_in[10];
    const float* ln1_g = (const float*)d_in[11];
    const float* ln1_b = (const float*)d_in[12];
    const float* W1    = (const float*)d_in[13];
    const float* b1    = (const float*)d_in[14];
    const float* W2    = (const float*)d_in[15];
    const float* b2    = (const float*)d_in[16];
    const float* ln2_g = (const float*)d_in[17];
    const float* ln2_b = (const float*)d_in[18];

    float* out = (float*)d_out;
    const size_t H_ELEMS = (size_t)CM * CD;          // 4096*512 hidden floats
    float* attnOut = out + H_ELEMS;                  // [B,L,H,S,S]

    // workspace carve-up (~81 MB total)
    char* w = (char*)d_ws;
    auto take = [&](size_t bytes) -> void* {
        void* p = (void*)w;
        w += (bytes + 255) & ~(size_t)255;
        return p;
    };
    bf16_t* x_bf   = (bf16_t*)take((size_t)CM * CIN * 2);
    bf16_t* We_bf  = (bf16_t*)take((size_t)CIN * CD * 2);
    bf16_t* Wq_bf  = (bf16_t*)take((size_t)CL * CD * CD * 2);
    bf16_t* Wk_bf  = (bf16_t*)take((size_t)CL * CD * CD * 2);
    bf16_t* Wv_bf  = (bf16_t*)take((size_t)CL * CD * CD * 2);
    bf16_t* Wo_bf  = (bf16_t*)take((size_t)CL * CD * CD * 2);
    bf16_t* W1_bf  = (bf16_t*)take((size_t)CL * CD * CDFF * 2);
    bf16_t* W2_bf  = (bf16_t*)take((size_t)CL * CDFF * CD * 2);
    float*  h_f32  = (float*) take((size_t)CM * CD * 4);
    bf16_t* h_bf   = (bf16_t*)take((size_t)CM * CD * 2);
    bf16_t* q_bf   = (bf16_t*)take((size_t)CM * CD * 2);
    bf16_t* k_bf   = (bf16_t*)take((size_t)CM * CD * 2);
    bf16_t* v_bf   = (bf16_t*)take((size_t)CM * CD * 2);
    bf16_t* ctx_bf = (bf16_t*)take((size_t)CM * CD * 2);
    float*  tmp    = (float*) take((size_t)CM * CD * 4);
    bf16_t* ff_bf  = (bf16_t*)take((size_t)CM * CDFF * 2);

    // --- convert inputs/weights to bf16 ---
    launch_cvt(x,  x_bf,  CM * CIN, stream);
    launch_cvt(We, We_bf, CIN * CD, stream);
    launch_cvt(Wq, Wq_bf, CL * CD * CD, stream);
    launch_cvt(Wk, Wk_bf, CL * CD * CD, stream);
    launch_cvt(Wv, Wv_bf, CL * CD * CD, stream);
    launch_cvt(Wo, Wo_bf, CL * CD * CD, stream);
    launch_cvt(W1, W1_bf, CL * CD * CDFF, stream);
    launch_cvt(W2, W2_bf, CL * CDFF * CD, stream);

    const dim3 blk(256);
    const dim3 gD (CM / TM, CD / TN);     // N=512  -> (64, 4)
    const dim3 gFF(CM / TM, CDFF / TN);   // N=2048 -> (64, 16)

    // --- embedding: h = x @ We + be + PE ---
    gemm_bf16<1,0,1,0,1,1><<<gD, blk, 0, stream>>>(
        x_bf, We_bf, be, nullptr, h_f32, h_bf, CM, CD, CIN);

    for (int l = 0; l < CL; ++l) {
        const bf16_t* wq = Wq_bf + (size_t)l * CD * CD;
        const bf16_t* wk = Wk_bf + (size_t)l * CD * CD;
        const bf16_t* wv = Wv_bf + (size_t)l * CD * CD;
        const bf16_t* wo = Wo_bf + (size_t)l * CD * CD;

        for (int pass = 0; pass < 2; ++pass) {
            // QKV projections (bf16 outputs only)
            gemm_bf16<1,0,0,0,0,1><<<gD, blk, 0, stream>>>(
                h_bf, wq, bq + l * CD, nullptr, nullptr, q_bf, CM, CD, CD);
            gemm_bf16<1,0,0,0,0,1><<<gD, blk, 0, stream>>>(
                h_bf, wk, bk + l * CD, nullptr, nullptr, k_bf, CM, CD, CD);
            gemm_bf16<1,0,0,0,0,1><<<gD, blk, 0, stream>>>(
                h_bf, wv, bv + l * CD, nullptr, nullptr, v_bf, CM, CD, CD);

            // banded attention (pass 2 materializes probabilities)
            attn_kernel<<<dim3(CS / 16, CB * CH), 32, 0, stream>>>(
                q_bf, k_bf, v_bf, ctx_bf, attnOut, l, pass == 1);

            // output projection + residual
            gemm_bf16<1,0,0,1,1,0><<<gD, blk, 0, stream>>>(
                ctx_bf, wo, bo + l * CD, h_f32, tmp, nullptr, CM, CD, CD);

            // LN1 -> new h (f32 + bf16)
            ln_kernel<<<CM, 256, 0, stream>>>(
                tmp, ln1_g + l * CD, ln1_b + l * CD, h_f32, h_bf);
        }

        // FFN: gelu(h @ W1 + b1) @ W2 + b2 + h, then LN2
        gemm_bf16<1,1,0,0,0,1><<<gFF, blk, 0, stream>>>(
            h_bf, W1_bf + (size_t)l * CD * CDFF, b1 + l * CDFF,
            nullptr, nullptr, ff_bf, CM, CDFF, CD);
        gemm_bf16<1,0,0,1,1,0><<<gD, blk, 0, stream>>>(
            ff_bf, W2_bf + (size_t)l * CDFF * CD, b2 + l * CD,
            h_f32, tmp, nullptr, CM, CD, CDFF);
        ln_kernel<<<CM, 256, 0, stream>>>(
            tmp, ln2_g + l * CD, ln2_b + l * CD, h_f32, h_bf);
    }

    // final hidden state -> d_out[0 : B*S*D]
    (void)hipMemcpyAsync(out, h_f32, H_ELEMS * sizeof(float),
                         hipMemcpyDeviceToDevice, stream);
}
